// SoftFullAttention_63024350102089
// MI455X (gfx1250) — matrix operand order repaired and checked by reading, not compiled, not run on hardware
//
#include <hip/hip_runtime.h>

#define B_ 4
#define C_ 32
#define H_ 256
#define W_ 256

typedef __attribute__((ext_vector_type(16))) _Float16 v16h;
typedef __attribute__((ext_vector_type(8)))  _Float16 v8h;
typedef __attribute__((ext_vector_type(8)))  float    v8f;

static __device__ __forceinline__ v16h cat8(v8h lo, v8h hi) {
  return __builtin_shufflevector(lo, hi, 0,1,2,3,4,5,6,7,8,9,10,11,12,13,14,15);
}

// ---------------------------------------------------------------------------
// Kernel 1: 1x1 convs (channel mixing). One thread per pixel (w = lane).
//   SVt/TVt : [B][W][H][C]  f16 (vertical branch, pixel-major, contiguous C)
//   SHt/THt : [B][H][W][C]  f16 (horizontal branch)
//   Tg16    : [B][C][H][W]  f16 copy of target
// ---------------------------------------------------------------------------
__global__ __launch_bounds__(256) void prep_kernel(
    const float* __restrict__ src, const float* __restrict__ tgt,
    const float* __restrict__ Pv,  const float* __restrict__ Ph,
    const float* __restrict__ w1v, const float* __restrict__ b1v,
    const float* __restrict__ w2v, const float* __restrict__ b2v,
    const float* __restrict__ w1h, const float* __restrict__ b1h,
    const float* __restrict__ w2h, const float* __restrict__ b2h,
    _Float16* __restrict__ SVt, _Float16* __restrict__ TVt,
    _Float16* __restrict__ SHt, _Float16* __restrict__ THt,
    _Float16* __restrict__ Tg16)
{
  __shared__ float4 lw1[256], lw2[256], lw3[256], lw4[256];
  const int tid = threadIdx.x;
  lw1[tid] = ((const float4*)w1v)[tid];
  lw2[tid] = ((const float4*)w2v)[tid];
  lw3[tid] = ((const float4*)w1h)[tid];
  lw4[tid] = ((const float4*)w2h)[tid];
  __syncthreads();

  const int h = blockIdx.x, b = blockIdx.y, w = tid;
  const float pv = Pv[h * W_ + w];
  const float ph = Ph[w * H_ + h];   // P_h is (W,H); P_h.T[h][w] = P_h[w][h]

  float asv[32], atv[32], ash[32], ath[32];
#pragma unroll
  for (int o = 0; o < 32; ++o) { asv[o]=b1v[o]; atv[o]=b2v[o]; ash[o]=b1h[o]; ath[o]=b2h[o]; }

  const size_t pix   = (size_t)h * W_ + w;
  const size_t bbase = (size_t)b * C_ * H_ * W_;
#pragma unroll
  for (int c4 = 0; c4 < 8; ++c4) {
    float xsv[4], xtv[4], xsh[4], xth[4];
#pragma unroll
    for (int k = 0; k < 4; ++k) {
      const int c = c4 * 4 + k;
      const size_t idx = bbase + (size_t)c * H_ * W_ + pix;
      float s = src[idx];
      float t = tgt[idx];
      Tg16[idx] = (_Float16)t;
      xsv[k] = s + pv; xtv[k] = t + pv; xsh[k] = s + ph; xth[k] = t + ph;
    }
#pragma unroll
    for (int o = 0; o < 32; ++o) {
      float4 a = lw1[o*8+c4], bq = lw2[o*8+c4], cq = lw3[o*8+c4], dq = lw4[o*8+c4];
      asv[o] += a.x*xsv[0] + a.y*xsv[1] + a.z*xsv[2] + a.w*xsv[3];
      atv[o] += bq.x*xtv[0] + bq.y*xtv[1] + bq.z*xtv[2] + bq.w*xtv[3];
      ash[o] += cq.x*xsh[0] + cq.y*xsh[1] + cq.z*xsh[2] + cq.w*xsh[3];
      ath[o] += dq.x*xth[0] + dq.y*xth[1] + dq.z*xth[2] + dq.w*xth[3];
    }
  }
  const size_t vi = (((size_t)b * W_ + w) * H_ + h) * C_;
  const size_t hi = (((size_t)b * H_ + h) * W_ + w) * C_;
#pragma unroll
  for (int o = 0; o < 32; ++o) {
    SVt[vi + o] = (_Float16)asv[o];
    TVt[vi + o] = (_Float16)atv[o];
    SHt[hi + o] = (_Float16)ash[o];
    THt[hi + o] = (_Float16)ath[o];
  }
}

// ---------------------------------------------------------------------------
// Kernel 2: target BCHW f32 -> TgT f16 [B][W][C][H]  (32x32 LDS tiles)
// ---------------------------------------------------------------------------
__global__ __launch_bounds__(256) void tpose_tgt_kernel(
    const float* __restrict__ tgt, _Float16* __restrict__ TgT)
{
  __shared__ float tile[32][33];
  const int tx = threadIdx.x, ty = threadIdx.y;
  const int ht = (blockIdx.x >> 3) * 32, wt = (blockIdx.x & 7) * 32;
  const int c = blockIdx.y, b = blockIdx.z;
  const size_t base = (size_t)(b * C_ + c) * H_;
#pragma unroll
  for (int r = 0; r < 4; ++r)
    tile[ty + 8*r][tx] = tgt[(base + ht + ty + 8*r) * W_ + wt + tx];
  __syncthreads();
#pragma unroll
  for (int r = 0; r < 4; ++r) {
    const int w = wt + ty + 8*r;
    TgT[(((size_t)b * W_ + w) * C_ + c) * H_ + ht + tx] = (_Float16)tile[tx][ty + 8*r];
  }
}

// ---------------------------------------------------------------------------
// Kernel 3: fused two-stage attention per (b, col) slice.
// Each wave owns FOUR 16-row M-tiles (a1[4], acc[8]) so the stage-1 B tiles
// and stage-2 T tiles are loaded ONCE per 32-wide K-chunk and reused 4x:
// 16 v_wmma per chunk, 8 global b128 loads per chunk.
//  A layout: lane m holds K 0..15 contiguous; lane m+16 holds K 16..31.
//  B layout: lane n = column, K 0..15 contiguous (lanes 16-31: K 16..31).
// ---------------------------------------------------------------------------
__global__ __launch_bounds__(128) void attn_kernel(
    const _Float16* __restrict__ Abase, const _Float16* __restrict__ Bbase,
    const _Float16* __restrict__ Tbase, float* __restrict__ Obase,
    size_t tSliceB, size_t tSliceC, size_t tRow)
{
  __shared__ _Float16 sc[4 * 16 * 32];
  const int s    = blockIdx.x;
  const int lane = threadIdx.x & 31;
  const int wv   = threadIdx.x >> 5;
  const int l15  = lane & 15;
  const int hiH  = lane >> 4;            // 0: lanes 0-15, 1: lanes 16-31

  const _Float16* A  = Abase + (size_t)s * (256 * 32);
  const _Float16* Bm = Bbase + (size_t)s * (256 * 32);
  const _Float16* T  = Tbase + (size_t)(s >> 8) * tSliceB + (size_t)(s & 255) * tSliceC;
  float*          O  = Obase + (size_t)s * (256 * 32);
  _Float16* mysc = sc + wv * (16 * 32);
  const int mrow = hiH * 8;
  const int mt0  = wv * 4;

  // Stage-1 A tiles for 4 M-tiles (loop invariant)
  v16h a1[4];
#pragma unroll
  for (int i = 0; i < 4; ++i) {
    const _Float16* ar = A + (size_t)(mt0 + i) * (16 * 32) + l15 * 32 + hiH * 16;
    a1[i] = *(const v16h*)ar;
  }
  v8f acc[8];
#pragma unroll
  for (int i = 0; i < 8; ++i) acc[i] = (v8f){};

  for (int g0 = 0; g0 < 256; g0 += 32) {
    // stage-1 B tiles: columns g0..g0+15 and g0+16..g0+31 (shared by 4 M-tiles)
    v16h b0 = *(const v16h*)(Bm + (size_t)(g0 + l15) * 32 + hiH * 16);
    v16h b1 = *(const v16h*)(Bm + (size_t)(g0 + 16 + l15) * 32 + hiH * 16);
    // stage-2 B tiles: column n = c, K = g contiguous along tRow rows
    v16h t0 = *(const v16h*)(T + (size_t)l15 * tRow + g0 + hiH * 16);
    v16h t1 = *(const v16h*)(T + (size_t)(16 + l15) * tRow + g0 + hiH * 16);

#pragma unroll
    for (int i = 0; i < 4; ++i) {
      v8f z = {};
      v8f s0 = __builtin_amdgcn_wmma_f32_16x16x32_f16(false, a1[i], false, b0, (short)0, z, false, false);
      v8f s1 = __builtin_amdgcn_wmma_f32_16x16x32_f16(false, a1[i], false, b1, (short)0, z, false, false);

      // D-layout (col-distributed) -> LDS row-major f16 16x32
#pragma unroll
      for (int j = 0; j < 8; ++j) {
        mysc[(mrow + j) * 32 + l15]      = (_Float16)s0[j];
        mysc[(mrow + j) * 32 + 16 + l15] = (_Float16)s1[j];
      }
      __builtin_amdgcn_wave_barrier();   // same-wave LDS is in-order; stop reordering

      // re-gather scores as stage-2 A tile (row m = l15, K = 32 g-values)
      const _Float16* srow = mysc + l15 * 32 + hiH * 16;
      v16h a2 = *(const v16h*)srow;
      __builtin_amdgcn_wave_barrier();   // loads are serviced before next i's stores

      acc[2*i]   = __builtin_amdgcn_wmma_f32_16x16x32_f16(false, a2, false, t0, (short)0, acc[2*i],   false, false);
      acc[2*i+1] = __builtin_amdgcn_wmma_f32_16x16x32_f16(false, a2, false, t1, (short)0, acc[2*i+1], false, false);
    }
  }

#pragma unroll
  for (int i = 0; i < 4; ++i) {
    float* orow = O + (size_t)(mt0 + i) * (16 * 32);
#pragma unroll
    for (int j = 0; j < 8; ++j) {
      orow[(mrow + j) * 32 + l15]      = acc[2*i][j];
      orow[(mrow + j) * 32 + 16 + l15] = acc[2*i+1][j];
    }
  }
}

// ---------------------------------------------------------------------------
// Kernel 4: pack [.,.,pix,C] f32 intermediates -> BCHW output (coalesced)
//   src index = b*baseMulB + h*hs + x*xs + c ; out channel = c0 + c, width = x
// ---------------------------------------------------------------------------
__global__ __launch_bounds__(256) void pack_kernel(
    const float* __restrict__ src, float* __restrict__ out,
    size_t baseMulB, size_t hs, size_t xs, int c0)
{
  __shared__ float tile[32][33];
  const int tx = threadIdx.x, ty = threadIdx.y;
  const int x0 = blockIdx.x * 32;
  const int h = blockIdx.y, b = blockIdx.z;
  const float* sb = src + (size_t)b * baseMulB + (size_t)h * hs;
#pragma unroll
  for (int r = 0; r < 4; ++r)
    tile[ty + 8*r][tx] = sb[(size_t)(x0 + ty + 8*r) * xs + tx];
  __syncthreads();
#pragma unroll
  for (int r = 0; r < 4; ++r) {
    const int c = ty + 8*r;
    out[(((size_t)b * 2 * C_ + c0 + c) * H_ + h) * W_ + x0 + tx] = tile[tx][c];
  }
}

// ---------------------------------------------------------------------------
extern "C" void kernel_launch(void* const* d_in, const int* in_sizes, int n_in,
                              void* d_out, int out_size, void* d_ws, size_t ws_size,
                              hipStream_t stream) {
  (void)in_sizes; (void)n_in; (void)out_size; (void)ws_size;
  const float* src = (const float*)d_in[0];
  const float* tgt = (const float*)d_in[1];
  const float* Pv  = (const float*)d_in[2];
  const float* Ph  = (const float*)d_in[3];
  const float* w1v = (const float*)d_in[4];
  const float* b1v = (const float*)d_in[5];
  const float* w2v = (const float*)d_in[6];
  const float* b2v = (const float*)d_in[7];
  const float* w1h = (const float*)d_in[8];
  const float* b1h = (const float*)d_in[9];
  const float* w2h = (const float*)d_in[10];
  const float* b2h = (const float*)d_in[11];
  float* out = (float*)d_out;

  char* ws = (char*)d_ws;
  const size_t N1 = (size_t)B_ * C_ * H_ * W_;     // 8,388,608 elements
  _Float16* SVt  = (_Float16*)(ws);                // [B][W][H][C] f16
  _Float16* TVt  = (_Float16*)(ws + 2 * N1);
  _Float16* SHt  = (_Float16*)(ws + 4 * N1);       // [B][H][W][C] f16
  _Float16* THt  = (_Float16*)(ws + 6 * N1);
  _Float16* TgT  = (_Float16*)(ws + 8 * N1);       // [B][W][C][H] f16
  _Float16* Tg16 = (_Float16*)(ws + 10 * N1);      // [B][C][H][W] f16
  float*    vertT = (float*)(ws + 12 * N1);        // [B][W][H][C] f32
  float*    horT  = (float*)(ws + 16 * N1);        // [B][H][W][C] f32

  prep_kernel<<<dim3(H_, B_), 256, 0, stream>>>(
      src, tgt, Pv, Ph, w1v, b1v, w2v, b2v, w1h, b1h, w2h, b2h,
      SVt, TVt, SHt, THt, Tg16);

  tpose_tgt_kernel<<<dim3(64, C_, B_), dim3(32, 8), 0, stream>>>(tgt, TgT);

  // vertical: slice s = b*W + w ; T = TgT[b][w][c][g], row stride H
  attn_kernel<<<B_ * W_, 128, 0, stream>>>(
      SVt, TVt, TgT, vertT,
      (size_t)W_ * C_ * H_, (size_t)C_ * H_, (size_t)H_);

  // horizontal: slice s = b*H + h ; T = Tg16[b][c][h][v], row stride H*W
  attn_kernel<<<B_ * H_, 128, 0, stream>>>(
      SHt, THt, Tg16, horT,
      (size_t)C_ * H_ * W_, (size_t)W_, (size_t)H_ * W_);

  // horizontal -> channels [0,32)
  pack_kernel<<<dim3(8, H_, B_), dim3(32, 8), 0, stream>>>(
      horT, out, (size_t)H_ * W_ * C_, (size_t)W_ * C_, (size_t)C_, 0);
  // vertical -> channels [32,64)
  pack_kernel<<<dim3(8, H_, B_), dim3(32, 8), 0, stream>>>(
      vertT, out, (size_t)W_ * H_ * C_, (size_t)C_, (size_t)H_ * C_, C_);
}
